// GAT_78151224918248
// MI455X (gfx1250) — matrix operand order repaired and checked by reading, not compiled, not run on hardware
//
#include <hip/hip_runtime.h>
#include <math.h>

typedef __attribute__((ext_vector_type(16))) _Float16 v16h;
typedef __attribute__((ext_vector_type(8)))  _Float16 h8;
typedef __attribute__((ext_vector_type(8)))  float    v8f;

#define WMMA_F16(a, b, c) \
  __builtin_amdgcn_wmma_f32_16x16x32_f16(false, (a), false, (b), (short)0, (c), false, false)

// concat two 16B LDS loads into one v16h fragment
__device__ __forceinline__ v16h ld_frag_2x8(const _Float16* p0, const _Float16* p1) {
  h8 lo = *(const h8*)p0;
  h8 hi = *(const h8*)p1;
  return __builtin_shufflevector(lo, hi, 0, 1, 2, 3, 4, 5, 6, 7,
                                 8, 9, 10, 11, 12, 13, 14, 15);
}

// ---------- order-preserving float<->uint encoding for atomic segment-max ----
__device__ __forceinline__ unsigned enc_f32(float f) {
  unsigned u = __float_as_uint(f);
  return (u & 0x80000000u) ? ~u : (u | 0x80000000u);
}
__device__ __forceinline__ float dec_f32(unsigned e) {
  return (e & 0x80000000u) ? __uint_as_float(e & 0x7FFFFFFFu) : __uint_as_float(~e);
}
#define ENC_NEG_INF 0x007FFFFFu  // enc(-inf)

__device__ __forceinline__ float leaky02(float x) { return x > 0.0f ? x : 0.2f * x; }

// =====================================================================
// Kernel 1: fused  nf0 = s*cos(feat0 @ Wrbf + b)  ->  h1 = nf0 @ W1
// One wave per 16-row M-tile; 8 waves per block.
// GEMM1: M16 x K64(pad of 62) x N64  (8 WMMAs)
// GEMM2: M16 x K64 x N16            (2 WMMAs)  via per-wave LDS transpose
// LDS weights stored column-major (contiguous K per column) -> ds_load_b128.
// All fragment loops fully unrolled so acc[][] stays in static registers.
// =====================================================================
__global__ __launch_bounds__(256) void rbf_conv1_kernel(
    const float* __restrict__ feat0,  // [N0,62]
    const float* __restrict__ Wrbf,   // [62,64]
    const float* __restrict__ brbf,   // [64]
    const float* __restrict__ W1,     // [64,16]
    float* __restrict__ h1,           // [N0,16]
    int n0)
{
  __shared__ __align__(32) _Float16 sW[64 * 64];        // [col][k] Wrbf, K padded 62->64
  __shared__ __align__(32) _Float16 sW1[16 * 64];       // [col][k] W1
  __shared__ __align__(32) _Float16 sStage[8][16 * 64]; // per-wave nf0 tile, row major

  const int t = threadIdx.x;
  for (int i = t; i < 64 * 64; i += 256) {
    int c = i >> 6, k = i & 63;
    sW[i] = (k < 62) ? (_Float16)Wrbf[k * 64 + c] : (_Float16)0.0f;
  }
  for (int i = t; i < 16 * 64; i += 256) {
    int c = i >> 6, k = i & 63;
    sW1[i] = (_Float16)W1[k * 16 + c];
  }
  __syncthreads();

  const int wave = t >> 5, lane = t & 31;
  const int tile = blockIdx.x * 8 + wave;
  const int row0 = tile * 16;
  if (row0 >= n0) return;

  const bool hi   = (lane & 16) != 0;
  const int  arow = row0 + (lane & 15);
  const int  lcol = lane & 15;
  const int  kb   = hi ? 16 : 0;   // B-fragment K base within a 32-K step
  const int  ab   = hi ? 16 : 0;   // A-fragment K base offset

  // speculative prefetch of the next block's A-tile (emits global_prefetch_b8)
  __builtin_prefetch(feat0 + (size_t)(row0 + 8 * 16) * 62, 0, 3);

  // ---- A fragments (contiguous k order used here: lanes 0-15 K{0..15}, lanes 16-31 K{16..31})
  v16h a0, a1;
#pragma unroll
  for (int i = 0; i < 16; ++i) {
    int koff = i + ab;
    int k0 = koff, k1 = 32 + koff;
    float f0 = (arow < n0 && k0 < 62) ? feat0[(size_t)arow * 62 + k0] : 0.0f;
    float f1 = (arow < n0 && k1 < 62) ? feat0[(size_t)arow * 62 + k1] : 0.0f;
    a0[i] = (_Float16)f0;
    a1[i] = (_Float16)f1;
  }

  // ---- GEMM1 over 4 N-tiles (B: contiguous 32B runs in column-major LDS)
  v8f acc[4];
#pragma unroll
  for (int nt = 0; nt < 4; ++nt) {
    const _Float16* bp = &sW[(nt * 16 + lcol) * 64];
    v16h b0 = *(const v16h*)(bp + kb);        // K-step 0
    v16h b1 = *(const v16h*)(bp + 32 + kb);   // K-step 1
    v8f c = {0, 0, 0, 0, 0, 0, 0, 0};
    c = WMMA_F16(a0, b0, c);
    c = WMMA_F16(a1, b1, c);
    acc[nt] = c;
  }

  // ---- RBF nonlinearity, stage as f16 tile (row major 16x64)
  const float scale = 0.17677669529663689f;  // sqrt(2/64)
  _Float16* stg = sStage[wave];
#pragma unroll
  for (int nt = 0; nt < 4; ++nt) {
    const int col = nt * 16 + lcol;
    const float bias = brbf[col];
#pragma unroll
    for (int r = 0; r < 8; ++r) {
      int m = r + (hi ? 8 : 0);
      stg[m * 64 + col] = (_Float16)(scale * cosf(acc[nt][r] + bias));
    }
  }

  // ---- GEMM2: h = nf0_tile @ W1  (K=64 -> two k-steps, N=16)
  // A fragment rows: two contiguous 8-half runs (16B aligned) per k-step
  const _Float16* ap = &stg[lcol * 64];
  v16h c0 = ld_frag_2x8(ap + ab,      ap + ab + 8);
  v16h c1 = ld_frag_2x8(ap + 32 + ab, ap + 40 + ab);
  const _Float16* dp = &sW1[lcol * 64];
  v16h d0 = *(const v16h*)(dp + kb);
  v16h d1 = *(const v16h*)(dp + 32 + kb);

  v8f acc2 = {0, 0, 0, 0, 0, 0, 0, 0};
  acc2 = WMMA_F16(c0, d0, acc2);
  acc2 = WMMA_F16(c1, d1, acc2);

#pragma unroll
  for (int r = 0; r < 8; ++r) {
    int row = row0 + r + (hi ? 8 : 0);
    if (row < n0) h1[(size_t)row * 16 + lcol] = acc2[r];
  }
}

// =====================================================================
// Kernel: conv2 GEMM  h2 = relu(num1/denom1 + b1) @ W2   [N0,16]x[16,64]
// K padded 16->32, one k-step, 4 N-tiles per wave.
// =====================================================================
__global__ __launch_bounds__(256) void conv2_gemm_kernel(
    const float* __restrict__ num1,   // [N0,16]
    const float* __restrict__ denom1, // [N0,2]
    const float* __restrict__ b1,     // [16]
    const float* __restrict__ W2,     // [16,64]
    float* __restrict__ h2,           // [N0,64]
    int n0)
{
  __shared__ __align__(32) _Float16 sW2[64 * 32];  // [col][k], K padded 16->32
  const int t = threadIdx.x;
  for (int i = t; i < 64 * 32; i += 256) {
    int c = i >> 5, k = i & 31;
    sW2[i] = (k < 16) ? (_Float16)W2[k * 64 + c] : (_Float16)0.0f;
  }
  __syncthreads();

  const int wave = t >> 5, lane = t & 31;
  const int row0 = (blockIdx.x * 8 + wave) * 16;
  if (row0 >= n0) return;

  const bool hi   = (lane & 16) != 0;
  const int  arow = row0 + (lane & 15);
  const int  lcol = lane & 15;
  const int  kb   = hi ? 16 : 0;

  __builtin_prefetch(num1 + (size_t)(row0 + 8 * 16) * 16, 0, 3);

  v16h a;
#pragma unroll
  for (int i = 0; i < 16; ++i) {
    int k = i + (hi ? 16 : 0);  // 0..31, real K < 16
    float v = 0.0f;
    if (arow < n0 && k < 16) {
      float d = denom1[(size_t)arow * 2 + (k >> 3)];        // per-head denom
      v = num1[(size_t)arow * 16 + k] / fmaxf(d, 1e-9f) + b1[k];
      v = fmaxf(v, 0.0f);                                   // relu (h0)
    }
    a[i] = (_Float16)v;
  }

#pragma unroll
  for (int nt = 0; nt < 4; ++nt) {
    v16h b = *(const v16h*)(&sW2[(nt * 16 + lcol) * 32] + kb);
    v8f c = {0, 0, 0, 0, 0, 0, 0, 0};
    c = WMMA_F16(a, b, c);
#pragma unroll
    for (int r = 0; r < 8; ++r) {
      int row = row0 + r + (hi ? 8 : 0);
      if (row < n0) h2[(size_t)row * 64 + nt * 16 + lcol] = c[r];
    }
  }
}

// ================= attention scores (el/er) =================
__global__ void scores16_kernel(const float* __restrict__ h1,
                                const float* __restrict__ al,
                                const float* __restrict__ ar,
                                float* __restrict__ el, float* __restrict__ er, int n0)
{
  int n = blockIdx.x * blockDim.x + threadIdx.x;
  if (n >= n0) return;
#pragma unroll
  for (int h = 0; h < 2; ++h) {
    float sl = 0.0f, sr = 0.0f;
#pragma unroll
    for (int f = 0; f < 8; ++f) {
      float v = h1[(size_t)n * 16 + h * 8 + f];
      sl += v * al[h * 8 + f];
      sr += v * ar[h * 8 + f];
    }
    el[n * 2 + h] = sl;
    er[n * 2 + h] = sr;
  }
}

__global__ void scores64_kernel(const float* __restrict__ h2,
                                const float* __restrict__ al,
                                const float* __restrict__ ar,
                                float* __restrict__ el, float* __restrict__ er, int n0)
{
  int n = blockIdx.x * blockDim.x + threadIdx.x;
  if (n >= n0) return;
  float sl = 0.0f, sr = 0.0f;
#pragma unroll
  for (int f = 0; f < 64; ++f) {
    float v = h2[(size_t)n * 64 + f];
    sl += v * al[f];
    sr += v * ar[f];
  }
  el[n] = sl;
  er[n] = sr;
}

// ================= edge passes: conv1 (2 heads x 8 feats) =================
__global__ void edge_max16_kernel(const int* __restrict__ src, const int* __restrict__ dst,
                                  const float* __restrict__ el, const float* __restrict__ er,
                                  unsigned* __restrict__ emax, int e0)
{
  int e = blockIdx.x * blockDim.x + threadIdx.x;
  if (e >= e0) return;
  int s = src[e], d = dst[e];
#pragma unroll
  for (int h = 0; h < 2; ++h) {
    float v = leaky02(el[s * 2 + h] + er[d * 2 + h]);
    atomicMax(&emax[d * 2 + h], enc_f32(v));
  }
}

__global__ void edge_accum16_kernel(const int* __restrict__ src, const int* __restrict__ dst,
                                    const float* __restrict__ el, const float* __restrict__ er,
                                    const unsigned* __restrict__ emax,
                                    const float* __restrict__ h1,
                                    float* __restrict__ denom, float* __restrict__ num, int e0)
{
  int e = blockIdx.x * blockDim.x + threadIdx.x;
  if (e >= e0) return;
  int s = src[e], d = dst[e];
#pragma unroll
  for (int h = 0; h < 2; ++h) {
    float v = leaky02(el[s * 2 + h] + er[d * 2 + h]);
    float a = expf(v - dec_f32(emax[d * 2 + h]));
    atomicAdd(&denom[d * 2 + h], a);
#pragma unroll
    for (int f = 0; f < 8; ++f)
      atomicAdd(&num[(size_t)d * 16 + h * 8 + f], a * h1[(size_t)s * 16 + h * 8 + f]);
  }
}

// ================= edge passes: conv2 (1 head x 64 feats) =================
__global__ void edge_max64_kernel(const int* __restrict__ src, const int* __restrict__ dst,
                                  const float* __restrict__ el, const float* __restrict__ er,
                                  unsigned* __restrict__ emax, int e0)
{
  int e = blockIdx.x * blockDim.x + threadIdx.x;
  if (e >= e0) return;
  float v = leaky02(el[src[e]] + er[dst[e]]);
  atomicMax(&emax[dst[e]], enc_f32(v));
}

__global__ void edge_accum64_kernel(const int* __restrict__ src, const int* __restrict__ dst,
                                    const float* __restrict__ el, const float* __restrict__ er,
                                    const unsigned* __restrict__ emax,
                                    const float* __restrict__ h2,
                                    float* __restrict__ denom, float* __restrict__ num, int e0)
{
  long long idx = (long long)blockIdx.x * blockDim.x + threadIdx.x;
  if (idx >= (long long)e0 * 64) return;
  int e = (int)(idx >> 6), f = (int)(idx & 63);
  int s = src[e], d = dst[e];
  float v = leaky02(el[s] + er[d]);
  float a = expf(v - dec_f32(emax[d]));
  if (f == 0) atomicAdd(&denom[d], a);
  atomicAdd(&num[(size_t)d * 64 + f], a * h2[(size_t)s * 64 + f]);
}

// ================= column-sum of out2 = num2/denom2 + b2 =================
__global__ __launch_bounds__(256) void colsum_kernel(
    const float* __restrict__ num2, const float* __restrict__ denom2,
    const float* __restrict__ b2, float* __restrict__ colsum, int n0)
{
  __shared__ float red[256];
  int t = threadIdx.x;
  int f = t & 63, g = t >> 6;
  float s = 0.0f;
  for (int n = blockIdx.x * 4 + g; n < n0; n += gridDim.x * 4)
    s += num2[(size_t)n * 64 + f] / fmaxf(denom2[n], 1e-9f) + b2[f];
  red[t] = s;
  __syncthreads();
  if (t < 64) atomicAdd(&colsum[f], red[t] + red[t + 64] + red[t + 128] + red[t + 192]);
}

// ================= dense tail =================
__global__ void tail_kernel(const float* __restrict__ colsum,
                            const float* __restrict__ fc1_w, const float* __restrict__ fc1_b,
                            const float* __restrict__ out_w, const float* __restrict__ out_b,
                            float* __restrict__ out, int n0)
{
  __shared__ float m[64];
  __shared__ float hfc[16];
  int t = threadIdx.x;
  if (t < 64) m[t] = fmaxf(colsum[t] / (float)n0, 0.0f);
  __syncthreads();
  if (t < 16) {
    float s = fc1_b[t];
#pragma unroll
    for (int f = 0; f < 64; ++f) s += m[f] * fc1_w[t * 64 + f];
    hfc[t] = fmaxf(s, 0.0f);
  }
  __syncthreads();
  if (t == 0) {
    float s = out_b[0];
#pragma unroll
    for (int j = 0; j < 16; ++j) s += hfc[j] * out_w[j];
    out[0] = s;
  }
}

// ================= accumulator init (every call; graph-safe) =================
__global__ void init_kernel(float* denom1, float* num1, unsigned* emax1,
                            float* denom2, float* num2, unsigned* emax2,
                            float* colsum, int n0)
{
  long long i = (long long)blockIdx.x * blockDim.x + threadIdx.x;
  if (i >= (long long)n0 * 64) return;
  num2[i] = 0.0f;
  if (i < (long long)n0 * 16) num1[i] = 0.0f;
  if (i < (long long)n0 * 2) { denom1[i] = 0.0f; emax1[i] = ENC_NEG_INF; }
  if (i < n0) { denom2[i] = 0.0f; emax2[i] = ENC_NEG_INF; }
  if (i < 64) colsum[i] = 0.0f;
}

// =====================================================================
extern "C" void kernel_launch(void* const* d_in, const int* in_sizes, int n_in,
                              void* d_out, int out_size, void* d_ws, size_t ws_size,
                              hipStream_t stream) {
  (void)in_sizes; (void)n_in; (void)out_size; (void)ws_size;
  const int N0 = 200000, E0 = 1000000;

  // g1 branch (inputs 0,1,2,6,7,8,11..18) is dead w.r.t. the output — skipped.
  const float* feat0 = (const float*)d_in[3];
  const int*   src0  = (const int*)d_in[4];
  const int*   dst0  = (const int*)d_in[5];
  const float* Wrbf0 = (const float*)d_in[9];
  const float* brbf0 = (const float*)d_in[10];
  const float* W1    = (const float*)d_in[19];  // g2c1_W  [64,16]
  const float* al1   = (const float*)d_in[20];  // g2c1_al [2,8]
  const float* ar1   = (const float*)d_in[21];
  const float* b1    = (const float*)d_in[22];  // [16]
  const float* W2    = (const float*)d_in[23];  // g2c2_W  [16,64]
  const float* al2   = (const float*)d_in[24];  // [64]
  const float* ar2   = (const float*)d_in[25];
  const float* b2    = (const float*)d_in[26];  // [64]
  const float* fc1_w = (const float*)d_in[27];  // [16,64]
  const float* fc1_b = (const float*)d_in[28];
  const float* out_w = (const float*)d_in[29];  // [1,16]
  const float* out_b = (const float*)d_in[30];

  // ---- workspace layout (all 4-byte elements) ----
  float* ws = (float*)d_ws;
  float*    h1     = ws; ws += (size_t)N0 * 16;
  float*    el1    = ws; ws += (size_t)N0 * 2;
  float*    er1    = ws; ws += (size_t)N0 * 2;
  unsigned* emax1  = (unsigned*)ws; ws += (size_t)N0 * 2;
  float*    denom1 = ws; ws += (size_t)N0 * 2;
  float*    num1   = ws; ws += (size_t)N0 * 16;
  float*    h2     = ws; ws += (size_t)N0 * 64;
  float*    el2    = ws; ws += (size_t)N0;
  float*    er2    = ws; ws += (size_t)N0;
  unsigned* emax2  = (unsigned*)ws; ws += (size_t)N0;
  float*    denom2 = ws; ws += (size_t)N0;
  float*    num2   = ws; ws += (size_t)N0 * 64;
  float*    colsum = ws; ws += 64;

  const int tiles  = (N0 + 15) / 16;         // 12500
  const int gblk   = (tiles + 7) / 8;        // 8 waves (M-tiles) per block

  init_kernel<<<((size_t)N0 * 64 + 255) / 256, 256, 0, stream>>>(
      denom1, num1, emax1, denom2, num2, emax2, colsum, N0);

  rbf_conv1_kernel<<<gblk, 256, 0, stream>>>(feat0, Wrbf0, brbf0, W1, h1, N0);
  scores16_kernel<<<(N0 + 255) / 256, 256, 0, stream>>>(h1, al1, ar1, el1, er1, N0);
  edge_max16_kernel<<<(E0 + 255) / 256, 256, 0, stream>>>(src0, dst0, el1, er1, emax1, E0);
  edge_accum16_kernel<<<(E0 + 255) / 256, 256, 0, stream>>>(
      src0, dst0, el1, er1, emax1, h1, denom1, num1, E0);

  conv2_gemm_kernel<<<gblk, 256, 0, stream>>>(num1, denom1, b1, W2, h2, N0);
  scores64_kernel<<<(N0 + 255) / 256, 256, 0, stream>>>(h2, al2, ar2, el2, er2, N0);
  edge_max64_kernel<<<(E0 + 255) / 256, 256, 0, stream>>>(src0, dst0, el2, er2, emax2, E0);
  edge_accum64_kernel<<<((size_t)E0 * 64 + 255) / 256, 256, 0, stream>>>(
      src0, dst0, el2, er2, emax2, h2, denom2, num2, E0);

  colsum_kernel<<<512, 256, 0, stream>>>(num2, denom2, b2, colsum, N0);
  tail_kernel<<<1, 64, 0, stream>>>(colsum, fc1_w, fc1_b, out_w, out_b, (float*)d_out, N0);
}
